// TransformerVMP_51015621542090
// MI455X (gfx1250) — matrix operand order repaired and checked by reading, not compiled, not run on hardware
//
#include <hip/hip_runtime.h>

// ---------------------------------------------------------------- constants
#define TB 8                      // batch
#define TT 512                    // sequence length (T_IN == T_OUT)
#define TD 512                    // model dim
#define NH 8                      // heads
#define DH 64                     // head dim
#define MTOT (TB*TT)              // 4096 rows in every activation matrix
#define MD ((size_t)MTOT*TD)      // elements per activation tensor

#define SCALE_A  (4.419417382415922e-02f)   // 1/sqrt(512)
#define SCALE_VA (1.0f/512.0f)              // 1/512

typedef __attribute__((ext_vector_type(16))) _Float16 v16h;
typedef __attribute__((ext_vector_type(8)))  float    v8f;
typedef _Float16 h16;

static __device__ __forceinline__ v8f wmma16(v16h a, v16h b, v8f c) {
  // D = A(16x32,f16) * B(32x16,f16) + C(16x16,f32)
  return __builtin_amdgcn_wmma_f32_16x16x32_f16(false, a, false, b, (short)0, c,
                                                false, false);
}

// f16 A-fragment (16x32): base pre-offset by 16*sel; one 16-half run.
static __device__ __forceinline__ v16h loadA(const h16* __restrict__ p) {
  v16h a;
#pragma unroll
  for (int e = 0; e < 8; ++e) {
    a[e]     = p[e];
    a[8 + e] = p[8 + e];
  }
  return a;
}
// f16 B-fragment (32x16): base pre-offset by 16*sel; one 16-half run.
static __device__ __forceinline__ v16h loadB(const h16* __restrict__ p) {
  v16h b;
#pragma unroll
  for (int e = 0; e < 16; ++e) b[e] = p[e];
  return b;
}

// ================================================================ prep passes
// weights: Wh = (h)W ; Ph = (h)(W^2 + vW) ; vWh = (h)vW      (TD x TD)
__global__ void __launch_bounds__(256) k_prep_w(
    const float* __restrict__ W, const float* __restrict__ vW,
    h16* __restrict__ Wh, h16* __restrict__ Ph, h16* __restrict__ vWh)
{
  const size_t i = (size_t)blockIdx.x * blockDim.x + threadIdx.x;
  const float w = W[i], vw = vW[i];
  Wh[i]  = (h16)w;
  Ph[i]  = (h16)(w*w + vw);
  vWh[i] = (h16)vw;
}

// activations: Xh = (h)Xm ; vXh = (h)Xv   (squares are formed in-register)
__global__ void __launch_bounds__(256) k_prep_x(
    const float* __restrict__ Xm, const float* __restrict__ Xv,
    h16* __restrict__ Xh, h16* __restrict__ vXh)
{
  const size_t i = (size_t)blockIdx.x * blockDim.x + threadIdx.x;
  Xh[i]  = (h16)Xm[i];
  vXh[i] = (h16)Xv[i];
}

// ----------------------------------------------------------------------------
// Shared VMP GEMM core: wave computes a 32x64 strip (2 M-tiles x 4 N-tiles).
// Three phases per K-step keep same-accumulator WMMAs far apart (no hazard
// NOPs) and hold only one B fragment live at a time.  All tile offsets are
// compile-time immediates off 5 walking base pointers.
// ----------------------------------------------------------------------------
#define VMP_KSTEP(pX, pVx, pW, pP, pVW, STRA, STRB)                            \
  {                                                                            \
    v16h aX[2], aV[2], aX2[2];                                                 \
    _Pragma("unroll") for (int rt = 0; rt < 2; ++rt) {                         \
      aX[rt]  = loadA(pX  + rt*16*STRA);                                       \
      aV[rt]  = loadA(pVx + rt*16*STRA);                                       \
      aX2[rt] = aX[rt] * aX[rt];                                               \
    }                                                                          \
    _Pragma("unroll") for (int t = 0; t < 4; ++t) {                            \
      const v16h bW = loadB(pW + t*16*STRB);                                   \
      _Pragma("unroll") for (int rt = 0; rt < 2; ++rt)                         \
        accM[rt][t] = wmma16(aX[rt], bW, accM[rt][t]);                         \
    }                                                                          \
    _Pragma("unroll") for (int t = 0; t < 4; ++t) {                            \
      const v16h bP = loadB(pP + t*16*STRB);                                   \
      _Pragma("unroll") for (int rt = 0; rt < 2; ++rt)                         \
        accV[rt][t] = wmma16(aV[rt], bP, accV[rt][t]);                         \
    }                                                                          \
    _Pragma("unroll") for (int t = 0; t < 4; ++t) {                            \
      const v16h bV = loadB(pVW + t*16*STRB);                                  \
      _Pragma("unroll") for (int rt = 0; rt < 2; ++rt)                         \
        accV[rt][t] = wmma16(aX2[rt], bV, accV[rt][t]);                        \
    }                                                                          \
    pX += 32; pVx += 32; pW += 32; pP += 32; pVW += 32;                        \
  }

// ======================================================= dense VMP GEMM (f32 out)
// Ym = X @ W^T (+b) ; Yv = vX @ (W^2+vW)^T + X^2 @ vW^T (+vb)
// grid = (TD/64, MTOT/128), block = 128 (4 waves); wave -> 32x64 strip.
__global__ void __launch_bounds__(128) k_gemm_f32out(
    const h16* __restrict__ Xh, const h16* __restrict__ vXh,
    const h16* __restrict__ Wh, const h16* __restrict__ Ph, const h16* __restrict__ vWh,
    const float* __restrict__ bias, const float* __restrict__ vbias,
    float* __restrict__ Ym, float* __restrict__ Yv)
{
  const int lane = threadIdx.x & 31;
  const int wave = threadIdx.x >> 5;
  const int l16  = lane & 15;
  const int sel  = lane >> 4;
  const int row0 = blockIdx.y * 128 + wave * 32;
  const int n0   = blockIdx.x * 64;

  v8f accM[2][4] = {}, accV[2][4] = {};
  const h16* pX  = Xh  + (size_t)(row0 + l16) * TD + 16*sel;
  const h16* pVx = vXh + (size_t)(row0 + l16) * TD + 16*sel;
  const h16* pW  = Wh  + (size_t)(n0 + l16) * TD + 16*sel;
  const h16* pP  = Ph  + (size_t)(n0 + l16) * TD + 16*sel;
  const h16* pVW = vWh + (size_t)(n0 + l16) * TD + 16*sel;

  for (int k0 = 0; k0 < TD; k0 += 32)
    VMP_KSTEP(pX, pVx, pW, pP, pVW, TD, TD)

#pragma unroll
  for (int t = 0; t < 4; ++t) {
    const int col = n0 + t*16 + l16;
    const float bm = bias  ? bias[col]  : 0.f;
    const float bv = vbias ? vbias[col] : 0.f;
#pragma unroll
    for (int rt = 0; rt < 2; ++rt)
#pragma unroll
      for (int r = 0; r < 8; ++r) {
        const size_t off = (size_t)(row0 + rt*16 + 8*sel + r) * TD + col;
        Ym[off] = accM[rt][t][r] + bm;
        Yv[off] = accV[rt][t][r] + bv;
      }
  }
}

// ===================================================== projection GEMM (f16 out)
// Emits f16 triple straight from f32 accumulators:
//   Mh = (h)m ; [Ph16 = (h)(m*m + v)] ; Vh = (h)v
// trans==0: row-major (MTOT x TD).  trans==1: head-transposed (b,h,n,t).
__global__ void __launch_bounds__(128) k_gemm_h(
    const h16* __restrict__ Xh, const h16* __restrict__ vXh,
    const h16* __restrict__ Wh, const h16* __restrict__ Ph, const h16* __restrict__ vWh,
    h16* __restrict__ Mh, h16* __restrict__ Ph16, h16* __restrict__ Vh, int trans)
{
  const int lane = threadIdx.x & 31;
  const int wave = threadIdx.x >> 5;
  const int l16  = lane & 15;
  const int sel  = lane >> 4;
  const int row0 = blockIdx.y * 128 + wave * 32;
  const int n0   = blockIdx.x * 64;

  v8f accM[2][4] = {}, accV[2][4] = {};
  const h16* pX  = Xh  + (size_t)(row0 + l16) * TD + 16*sel;
  const h16* pVx = vXh + (size_t)(row0 + l16) * TD + 16*sel;
  const h16* pW  = Wh  + (size_t)(n0 + l16) * TD + 16*sel;
  const h16* pP  = Ph  + (size_t)(n0 + l16) * TD + 16*sel;
  const h16* pVW = vWh + (size_t)(n0 + l16) * TD + 16*sel;

  for (int k0 = 0; k0 < TD; k0 += 32)
    VMP_KSTEP(pX, pVx, pW, pP, pVW, TD, TD)

#pragma unroll
  for (int t = 0; t < 4; ++t) {
    const int col = n0 + t*16 + l16;
#pragma unroll
    for (int rt = 0; rt < 2; ++rt)
#pragma unroll
      for (int r = 0; r < 8; ++r) {
        const int row = row0 + rt*16 + 8*sel + r;
        const float m = accM[rt][t][r], v = accV[rt][t][r];
        size_t off;
        if (!trans) {
          off = (size_t)row * TD + col;
        } else {
          const int bb = row >> 9, tt = row & (TT - 1);
          const int hh = col >> 6, nn = col & (DH - 1);
          off = ((size_t)(bb*NH + hh)*DH + nn) * TT + tt;
        }
        Mh[off] = (h16)m;
        if (Ph16) Ph16[off] = (h16)(m*m + v);
        Vh[off] = (h16)v;
      }
  }
}

// ============================================================ attention scores
// per (b,h): a = q@k^T / sqrt(D);  va = (vq@(k^2+vk)^T + q^2@vk^T) / D
// grid = (TT/64, TT/128, TB*NH), block = 128; wave -> 32x64 strip, K = 64.
__global__ void __launch_bounds__(128) k_attn_scores(
    float* __restrict__ A, float* __restrict__ VA,
    const h16* __restrict__ Qh, const h16* __restrict__ vQh,
    const h16* __restrict__ Kh, const h16* __restrict__ Kp, const h16* __restrict__ vKh,
    int masking)
{
  const int lane = threadIdx.x & 31;
  const int wave = threadIdx.x >> 5;
  const int l16  = lane & 15;
  const int sel  = lane >> 4;
  const int bh = blockIdx.z, b = bh >> 3, h = bh & 7;
  const int row0 = blockIdx.y * 128 + wave * 32;
  const int n0   = blockIdx.x * 64;

  v8f accM[2][4] = {}, accV[2][4] = {};
  const h16* pX  = Qh  + (size_t)(b*TT + row0 + l16) * TD + h*DH + 16*sel;
  const h16* pVx = vQh + (size_t)(b*TT + row0 + l16) * TD + h*DH + 16*sel;
  const h16* pW  = Kh  + (size_t)(b*TT + n0 + l16) * TD + h*DH + 16*sel;
  const h16* pP  = Kp  + (size_t)(b*TT + n0 + l16) * TD + h*DH + 16*sel;
  const h16* pVW = vKh + (size_t)(b*TT + n0 + l16) * TD + h*DH + 16*sel;

#pragma unroll
  for (int k0 = 0; k0 < DH; k0 += 32)
    VMP_KSTEP(pX, pVx, pW, pP, pVW, TD, TD)

#pragma unroll
  for (int t = 0; t < 4; ++t)
#pragma unroll
    for (int rt = 0; rt < 2; ++rt)
#pragma unroll
      for (int r = 0; r < 8; ++r) {
        const int row = row0 + rt*16 + 8*sel + r;
        const int col = n0 + t*16 + l16;
        float m = accM[rt][t][r] * SCALE_A;
        float v = accV[rt][t][r] * SCALE_VA;
        if (masking && col > row) m = -1e9f;   // reference masks mean only
        const size_t off = ((size_t)bh*TT + row) * TT + col;
        A[off]  = m;
        VA[off] = v;
      }
}

// ============================================================== softmax (VMP)
// s = softmax(a); emits f16 rows:  Sh = (h)s ; VSh = (h)((s*(1-s))^2 * va)
__global__ void __launch_bounds__(256) k_softmax_vmp(
    const float* __restrict__ A, const float* __restrict__ VA,
    h16* __restrict__ Sh, h16* __restrict__ VSh)
{
  __shared__ float red[256];
  const size_t base = (size_t)blockIdx.x * TT;
  const int tid = threadIdx.x;
  const float x0 = A[base + tid], x1 = A[base + tid + 256];
  red[tid] = fmaxf(x0, x1); __syncthreads();
  for (int s = 128; s > 0; s >>= 1) { if (tid < s) red[tid] = fmaxf(red[tid], red[tid+s]); __syncthreads(); }
  const float mx = red[0]; __syncthreads();
  const float e0 = expf(x0 - mx), e1 = expf(x1 - mx);
  red[tid] = e0 + e1; __syncthreads();
  for (int s = 128; s > 0; s >>= 1) { if (tid < s) red[tid] += red[tid+s]; __syncthreads(); }
  const float inv = 1.f / red[0];
  const float s0 = e0*inv, s1 = e1*inv;
  const float g0 = s0*(1.f - s0), g1 = s1*(1.f - s1);
  Sh[base + tid]        = (h16)s0;
  Sh[base + tid + 256]  = (h16)s1;
  VSh[base + tid]       = (h16)(g0*g0*VA[base + tid]);
  VSh[base + tid + 256] = (h16)(g1*g1*VA[base + tid + 256]);
}

// ============================================================ attention output
// per (b,h): o = s@v ; vo = vs@(v^2+vv) + s^2@vv  -> head slice of f32 output.
// V operands are head-transposed (contiguous K-runs).  grid = (1, TT/128, TB*NH).
__global__ void __launch_bounds__(128) k_attn_out(
    float* __restrict__ Om, float* __restrict__ Ov,
    const h16* __restrict__ Sh, const h16* __restrict__ VSh,
    const h16* __restrict__ Vth, const h16* __restrict__ Vtp, const h16* __restrict__ Vtv)
{
  const int lane = threadIdx.x & 31;
  const int wave = threadIdx.x >> 5;
  const int l16  = lane & 15;
  const int sel  = lane >> 4;
  const int bh = blockIdx.z, b = bh >> 3, h = bh & 7;
  const int row0 = blockIdx.y * 128 + wave * 32;

  v8f accM[2][4] = {}, accV[2][4] = {};
  const h16* pX  = Sh  + ((size_t)bh*TT + row0 + l16) * TT + 16*sel;
  const h16* pVx = VSh + ((size_t)bh*TT + row0 + l16) * TT + 16*sel;
  const h16* pW  = Vth + ((size_t)(b*NH + h)*DH + l16) * TT + 16*sel;
  const h16* pP  = Vtp + ((size_t)(b*NH + h)*DH + l16) * TT + 16*sel;
  const h16* pVW = Vtv + ((size_t)(b*NH + h)*DH + l16) * TT + 16*sel;

  for (int k0 = 0; k0 < TT; k0 += 32)
    VMP_KSTEP(pX, pVx, pW, pP, pVW, TT, TT)

#pragma unroll
  for (int t = 0; t < 4; ++t)
#pragma unroll
    for (int rt = 0; rt < 2; ++rt)
#pragma unroll
      for (int r = 0; r < 8; ++r) {
        const size_t off = (size_t)(b*TT + row0 + rt*16 + 8*sel + r) * TD
                         + h*DH + t*16 + l16;
        Om[off] = accM[rt][t][r];
        Ov[off] = accV[rt][t][r];
      }
}

// ============================================================== layernorm VMP
__global__ void __launch_bounds__(256) k_layernorm_vmp(
    float* __restrict__ Xm, float* __restrict__ Xv,
    const float* __restrict__ gw, const float* __restrict__ bw,
    const float* __restrict__ vg, const float* __restrict__ vb)
{
  __shared__ float red[256];
  const size_t base = (size_t)blockIdx.x * TD;
  const int tid = threadIdx.x;
  const float x0 = Xm[base + tid], x1 = Xm[base + tid + 256];
  const float v0 = Xv[base + tid], v1 = Xv[base + tid + 256];
  red[tid] = x0 + x1; __syncthreads();
  for (int s = 128; s > 0; s >>= 1) { if (tid < s) red[tid] += red[tid+s]; __syncthreads(); }
  const float mu = red[0] * (1.f/TD); __syncthreads();
  red[tid] = x0*x0 + x1*x1; __syncthreads();
  for (int s = 128; s > 0; s >>= 1) { if (tid < s) red[tid] += red[tid+s]; __syncthreads(); }
  const float sig2 = red[0] * (1.f/TD) - mu*mu;
  const float inv  = 1.f / (sig2 + 1e-6f);
  const float rstd = sqrtf(inv);
  const float xh0 = (x0 - mu) * rstd, xh1 = (x1 - mu) * rstd;
  const float g0 = gw[tid], g1 = gw[tid + 256];
  Xm[base + tid]       = g0*xh0 + bw[tid];
  Xm[base + tid + 256] = g1*xh1 + bw[tid + 256];
  Xv[base + tid]       = g0*g0*v0*inv + vg[tid]*xh0*xh0 + vb[tid];
  Xv[base + tid + 256] = g1*g1*v1*inv + vg[tid + 256]*xh1*xh1 + vb[tid + 256];
}

// ============================================================== elementwise
__global__ void __launch_bounds__(256) k_embed_enc(
    const float* __restrict__ x, float* __restrict__ cm, float* __restrict__ cv)
{
  const size_t i = (size_t)blockIdx.x * blockDim.x + threadIdx.x;
  const int d = (int)(i & (TD - 1));
  const int t = (int)((i >> 9) & (TT - 1));
  const double ang = (double)t * exp(-log(10000.0) * (double)(d & ~1) / (double)TD);
  const float pe = (d & 1) ? (float)cos(ang) : (float)sin(ang);
  cm[i] = x[i] + pe;
  cv[i] = 0.f;
}

__global__ void __launch_bounds__(256) k_embed_dec(
    const float* __restrict__ y, float* __restrict__ cm, float* __restrict__ cv)
{
  const size_t i = (size_t)blockIdx.x * blockDim.x + threadIdx.x;
  const int d = (int)(i & (TD - 1));
  const int t = (int)((i >> 9) & (TT - 1));
  const int b = (int)(i >> 18);
  const double ang = (double)t * exp(-log(10000.0) * (double)(d & ~1) / (double)TD);
  const float pe = (d & 1) ? (float)cos(ang) : (float)sin(ang);
  const float base = (t == 0) ? 0.5f : y[((size_t)b*(TT-1) + (t-1))*TD + d];
  cm[i] = base + pe;
  cv[i] = 0.f;
}

__global__ void __launch_bounds__(256) k_residual(
    float* __restrict__ Xm, float* __restrict__ Xv,
    const float* __restrict__ Am, const float* __restrict__ Av)
{
  const size_t i = (size_t)blockIdx.x * blockDim.x + threadIdx.x;
  Xm[i] += Am[i];
  Xv[i] += Av[i];
}

__global__ void __launch_bounds__(256) k_relu_residual(
    float* __restrict__ Xm, float* __restrict__ Xv,
    const float* __restrict__ Am, const float* __restrict__ Av)
{
  const size_t i = (size_t)blockIdx.x * blockDim.x + threadIdx.x;
  const float m = Am[i];
  const float g = (m > 0.f) ? 1.f : 0.f;
  Xm[i] += m * g;
  Xv[i] += Av[i] * g;
}

__global__ void __launch_bounds__(256) k_sigmoid_out(
    const float* __restrict__ Xm, const float* __restrict__ Xv, float* __restrict__ out)
{
  const size_t i = (size_t)blockIdx.x * blockDim.x + threadIdx.x;
  const float s = 1.f / (1.f + expf(-Xm[i]));
  const float g = s * (1.f - s);
  out[i]      = s;
  out[MD + i] = g*g*Xv[i];
}

// ================================================================= launcher
extern "C" void kernel_launch(void* const* d_in, const int* in_sizes, int n_in,
                              void* d_out, int out_size, void* d_ws, size_t ws_size,
                              hipStream_t stream)
{
  const float* x = (const float*)d_in[0];
  const float* y = (const float*)d_in[1];
  // params flattened in jax-tree (sorted-key) order; index map per analysis.
  auto prm = [&](int i) { return (const float*)d_in[2 + i]; };

  // bump allocator over d_ws
  size_t off = 0;
  auto fAlloc = [&](size_t elems) { float* p = (float*)((char*)d_ws + off); off += elems*4; return p; };
  auto hAlloc = [&](size_t elems) { h16*   p = (h16*)  ((char*)d_ws + off); off += elems*2; return p; };

  float* cur_m = fAlloc(MD);  float* cur_v = fAlloc(MD);
  float* tmp_m = fAlloc(MD);  float* tmp_v = fAlloc(MD);
  float* abuf  = fAlloc(8*MD);
  float* vabuf = fAlloc(8*MD);
  h16* xh  = hAlloc(MD);  h16* vxh = hAlloc(MD);
  h16* qh  = hAlloc(MD);  h16* vqh = hAlloc(MD);
  h16* kh  = hAlloc(MD);  h16* kp  = hAlloc(MD);  h16* vkh = hAlloc(MD);
  h16* vth = hAlloc(MD);  h16* vtp = hAlloc(MD);  h16* vtv = hAlloc(MD);
  h16* ekh = hAlloc(MD);  h16* ekp = hAlloc(MD);  h16* evk = hAlloc(MD);
  h16* evh = hAlloc(MD);  h16* evp = hAlloc(MD);  h16* evv = hAlloc(MD);
  h16* sh  = hAlloc(8*MD);
  h16* vsh = hAlloc(8*MD);
  h16* wh  = hAlloc((size_t)TD*TD);
  h16* ph  = hAlloc((size_t)TD*TD);
  h16* vwh = hAlloc((size_t)TD*TD);

  const dim3 gGemm(TD/64, MTOT/128);
  const dim3 gScore(TT/64, TT/128, TB*NH);
  const dim3 gAOut(1, TT/128, TB*NH);
  const int  EW = (int)(MD / 256);

  auto prepX = [&]() {
    k_prep_x<<<EW, 256, 0, stream>>>(cur_m, cur_v, xh, vxh);
  };
  auto prepW = [&](int wi, int vwi) {
    k_prep_w<<<TD*TD/256, 256, 0, stream>>>(prm(wi), prm(vwi), wh, ph, vwh);
  };
  auto gemmF = [&](int wi, int vwi, int bi, int vbi, float* Ym, float* Yv) {
    prepW(wi, vwi);
    k_gemm_f32out<<<gGemm, 128, 0, stream>>>(xh, vxh, wh, ph, vwh,
        bi >= 0 ? prm(bi) : nullptr, vbi >= 0 ? prm(vbi) : nullptr, Ym, Yv);
  };
  auto gemmH = [&](int wi, int vwi, h16* Mh, h16* Pout, h16* Vh, int trans) {
    prepW(wi, vwi);
    k_gemm_h<<<gGemm, 128, 0, stream>>>(xh, vxh, wh, ph, vwh, Mh, Pout, Vh, trans);
  };
  auto ln = [&](int base) {   // _ln leaves: [b,g,vb,vg]
    k_layernorm_vmp<<<MTOT, 256, 0, stream>>>(cur_m, cur_v,
        prm(base+1), prm(base+0), prm(base+3), prm(base+2));
  };
  auto attn = [&](const h16* Kh_, const h16* Kp_, const h16* vK_,
                  const h16* Vt_, const h16* Vp_, const h16* Vv_, int masking) {
    k_attn_scores<<<gScore, 128, 0, stream>>>(abuf, vabuf, qh, vqh, Kh_, Kp_, vK_, masking);
    k_softmax_vmp<<<TB*NH*TT, 256, 0, stream>>>(abuf, vabuf, sh, vsh);
    k_attn_out<<<gAOut, 128, 0, stream>>>(tmp_m, tmp_v, sh, vsh, Vt_, Vp_, Vv_);
    k_residual<<<EW, 256, 0, stream>>>(cur_m, cur_v, tmp_m, tmp_v);
  };

  // ------------------------------- encoder -------------------------------
  k_embed_enc<<<EW, 256, 0, stream>>>(x, cur_m, cur_v);
  for (int i = 0; i < 2; ++i) {
    if (i) ln(92 + 0);                        // enc.norm1[0]
    prepX();
    const int a0 = 68 + 6*i;                  // [k.W,k.vW,q.W,q.vW,v.W,v.vW]
    gemmH(a0+2, a0+3, qh,  nullptr, vqh, 0);
    gemmH(a0+0, a0+1, kh,  kp,      vkh, 0);
    gemmH(a0+4, a0+5, vth, vtp,     vtv, 1);
    attn(kh, kp, vkh, vth, vtp, vtv, 0);
    ln(100 + 4*i);                            // enc.norm2[i]
    prepX();
    const int f0 = 80 + 4*i;                  // [W,b,vW,vb]
    gemmF(f0+0, f0+2, f0+1, f0+3, tmp_m, tmp_v);
    k_relu_residual<<<EW, 256, 0, stream>>>(cur_m, cur_v, tmp_m, tmp_v);
  }
  ln(92 + 4);                                 // enc.norm1[1]
  prepX();
  gemmH(88, 89, ekh, ekp, evk, 0);            // final.k (row-major triple)
  gemmH(90, 91, evh, evp, evv, 1);            // final.v (head-transposed triple)

  // ------------------------------- decoder -------------------------------
  k_embed_dec<<<EW, 256, 0, stream>>>(y, cur_m, cur_v);
  for (int i = 0; i < 3; ++i) {
    if (i) ln(18 + 4*(i-1));                  // dec.norm1[i-1]
    prepX();
    const int s0 = 50 + 6*i;                  // self [k,q,v]
    gemmH(s0+2, s0+3, qh,  nullptr, vqh, 0);
    gemmH(s0+0, s0+1, kh,  kp,      vkh, 0);
    gemmH(s0+4, s0+5, vth, vtp,     vtv, 1);
    attn(kh, kp, vkh, vth, vtp, vtv, 1);      // masked self-attention
    ln(26 + 4*i);                             // dec.norm2[i]
    prepX();
    gemmH(0 + 2*i, 1 + 2*i, qh, nullptr, vqh, 0);   // cross projection
    attn(ekh, ekp, evk, evh, evp, evv, 1);    // cross-attn (reference masks it too)
    ln(38 + 4*i);                             // dec.norm3[i]
    prepX();
    const int f0 = 6 + 4*i;
    gemmF(f0+0, f0+2, f0+1, f0+3, tmp_m, tmp_v);
    k_relu_residual<<<EW, 256, 0, stream>>>(cur_m, cur_v, tmp_m, tmp_v);
  }
  k_sigmoid_out<<<EW, 256, 0, stream>>>(cur_m, cur_v, (float*)d_out);

  (void)in_sizes; (void)n_in; (void)out_size; (void)ws_size;
}
